// MultiHeadAttention_45655502357196
// MI455X (gfx1250) — compile-verified, matrix operand order checked by reading, not run on hardware
//
#include <hip/hip_runtime.h>
#include <hip/hip_bf16.h>

// ---------------------------------------------------------------------------
// Problem constants (from reference): B=8, S=1024, DM=1024, H=16, DK=DV=64
// ---------------------------------------------------------------------------
#define BATCH 8
#define SEQ   1024
#define DMODEL 1024
#define NHEAD 16
#define HDIM  64

typedef __attribute__((ext_vector_type(16))) __bf16 bf16x16;
typedef __attribute__((ext_vector_type(8)))  __bf16 bf16x8;
typedef __attribute__((ext_vector_type(8)))  float  v8f;

// ---------------------------------------------------------------------------
// LDS fragment loaders (CDNA5 ISA 7.12.2 layouts, wave32).
// A (16x32 bf16, M x K): lane L -> row = rowBase + (L&15); the 16 K values are
// two contiguous 8-element (16B) runs at k = 8*(L>>4) and 16 + 8*(L>>4).
// B fragment (32x16, K x N) mirrors A with lane -> column, so it is read as
// rows of B^T staged in LDS with the identical routine.
// LDS tiles are row-padded (40 bf16 / 36 f32 per 32-element row) so the 16
// lanes of each half-wave land in 16 disjoint 4-bank groups.
// ---------------------------------------------------------------------------
__device__ __forceinline__ bf16x16 lds_frag(const __bf16* tile, int rowBase, int lane) {
  const int g = lane >> 4;
  const int r = lane & 15;
  const __bf16* p = tile + (rowBase + r) * 40 + (g << 3);
  bf16x8 c0 = *(const bf16x8*)(p);
  bf16x8 c1 = *(const bf16x8*)(p + 16);
  bf16x16 f;
#pragma unroll
  for (int j = 0; j < 8; ++j) { f[j] = c0[j]; f[j + 8] = c1[j]; }
  return f;
}

__device__ __forceinline__ bf16x16 lds_frag(const float* tile, int rowBase, int lane) {
  const int g = lane >> 4;
  const int r = lane & 15;
  const float* p = tile + (rowBase + r) * 36 + (g << 3);
  bf16x16 f;
#pragma unroll
  for (int j = 0; j < 8; ++j) {
    f[j]     = (__bf16)p[j];
    f[j + 8] = (__bf16)p[16 + j];
  }
  return f;
}

// ---------------------------------------------------------------------------
// Generic WMMA GEMM with async-to-LDS double-buffered staging:
//   C[z] = epilogue( A[z] (MxK) * Bt[z]^T (KxN) ),  Bt pre-transposed (N x K).
// Block = 128 threads (4 waves); block tile 64x64; BK = 32.
// Per K step: tiles are DMA'd to LDS with global_load_async_to_lds_b128
// (ASYNCcnt), next tile's DMA overlaps current tile's 16 v_wmma ops.
// EPI: 0 = f32 out (+bias), 1 = bf16 out (+bias),
//      2 = scores: f32 out = alpha*acc, mask==0 -> -1e9 (batch z = b*NHEAD+h)
// ---------------------------------------------------------------------------
template <typename AT, int EPI>
__global__ __launch_bounds__(128) void gemm_wmma_kernel(
    const AT* __restrict__ A, const __bf16* __restrict__ Bt, void* __restrict__ Cv,
    const float* __restrict__ bias, const int* __restrict__ mask,
    int K, int lda, int ldb, int ldc,
    long sA_, long sB_, long sC_, float alpha) {
  constexpr int BK    = 32;
  constexpr int PAD_A = (sizeof(AT) == 2) ? 40 : 36;       // elements per LDS row
  constexpr int CPR_A = (int)(BK * sizeof(AT)) / 16;       // 16B chunks per row
  constexpr int NA    = (64 * CPR_A) / 128;                // A chunks per thread
  constexpr int NB    = 2;                                 // B chunks per thread
  constexpr int NOUT  = NA + NB;                           // async ops per issue

  __shared__ AT     shA[2][64 * PAD_A];
  __shared__ __bf16 shB[2][64 * 40];

  const int z    = blockIdx.z;
  const AT*     Ab = A  + (size_t)z * sA_;
  const __bf16* Bb = Bt + (size_t)z * sB_;
  const int tid  = threadIdx.x;
  const int lane = tid & 31;
  const int wave = tid >> 5;
  const int blockM = blockIdx.y * 64;
  const int blockN = blockIdx.x * 64;
  const int m0 = blockM + wave * 16;
  const int n0 = blockN;

  const unsigned sAb0 = (unsigned)(unsigned long long)&shA[0][0];
  const unsigned sAb1 = (unsigned)(unsigned long long)&shA[1][0];
  const unsigned sBb0 = (unsigned)(unsigned long long)&shB[0][0];
  const unsigned sBb1 = (unsigned)(unsigned long long)&shB[1][0];

  // Issue async DMA of the 64x32 A tile and 64x32 B^T tile for one K step.
  auto issue = [&](int buf, int kBase) {
    const char* gAk = (const char*)Ab + (size_t)kBase * sizeof(AT);
    const unsigned dA = buf ? sAb1 : sAb0;
#pragma unroll
    for (int i = 0; i < NA; ++i) {
      const int c = tid + i * 128;
      const int row = c / CPR_A, sub = c % CPR_A;
      const unsigned long long src = (unsigned long long)(
          gAk + (size_t)(blockM + row) * lda * sizeof(AT) + sub * 16);
      const unsigned dst = dA + row * (PAD_A * (int)sizeof(AT)) + sub * 16;
      asm volatile("global_load_async_to_lds_b128 %0, %1, off"
                   :: "v"(dst), "v"(src) : "memory");
    }
    const char* gBk = (const char*)Bb + (size_t)kBase * 2;
    const unsigned dB = buf ? sBb1 : sBb0;
#pragma unroll
    for (int i = 0; i < NB; ++i) {
      const int c = tid + i * 128;
      const int row = c >> 2, sub = c & 3;
      const unsigned long long src = (unsigned long long)(
          gBk + (size_t)(blockN + row) * ldb * 2 + sub * 16);
      const unsigned dst = dB + row * 80 + sub * 16;
      asm volatile("global_load_async_to_lds_b128 %0, %1, off"
                   :: "v"(dst), "v"(src) : "memory");
    }
  };

  v8f acc[4] = {};
  const int kc = K / BK;

  issue(0, 0);  // prologue

  for (int ks = 0; ks < kc; ++ks) {
    const int cur = ks & 1;
    const bool more = (ks + 1) < kc;
    if (more) issue(cur ^ 1, (ks + 1) * BK);  // overlap next tile's DMA

    // Async loads complete in order: allowing NOUT outstanding means the
    // current buffer's copies have landed.
    if (more) {
      if (NOUT == 4) asm volatile("s_wait_asynccnt 0x4" ::: "memory");
      else           asm volatile("s_wait_asynccnt 0x6" ::: "memory");
    } else {
      asm volatile("s_wait_asynccnt 0x0" ::: "memory");
    }
    __syncthreads();  // current tile visible to all waves

    const AT*     tA = &shA[cur][0];
    const __bf16* tB = &shB[cur][0];
    bf16x16 af = lds_frag(tA, wave * 16, lane);
#pragma unroll
    for (int t = 0; t < 4; ++t) {
      bf16x16 bf = lds_frag(tB, t * 16, lane);
      acc[t] = __builtin_amdgcn_wmma_f32_16x16x32_bf16(
          false, af, false, bf, (short)0, acc[t], false, false);
    }
    __syncthreads();  // all waves done reading before this buffer is re-DMA'd
  }

  // Epilogue. C/D layout: VGPR vr -> row vr + 8*(lane>=16), col = lane&15.
  const int r = lane & 15;
  const int g = lane >> 4;
  const size_t cb = (size_t)z * sC_;
#pragma unroll
  for (int t = 0; t < 4; ++t) {
#pragma unroll
    for (int vr = 0; vr < 8; ++vr) {
      const int rowi = m0 + vr + g * 8;
      const int col  = n0 + t * 16 + r;
      float v = acc[t][vr];
      if (EPI == 2) {
        v *= alpha;
        const int b = z >> 4;  // z = b*NHEAD + h, NHEAD = 16
        const size_t mi = (((size_t)b * SEQ + rowi) * SEQ) + col;
        if (mask[mi] == 0) v = -1e9f;
        ((float*)Cv)[cb + (size_t)rowi * ldc + col] = v;
      } else if (EPI == 1) {
        if (bias) v += bias[col];
        ((__bf16*)Cv)[cb + (size_t)rowi * ldc + col] = (__bf16)v;
      } else {
        if (bias) v += bias[col];
        ((float*)Cv)[cb + (size_t)rowi * ldc + col] = v;
      }
    }
  }
}

// ---------------------------------------------------------------------------
// Row softmax over masked scores (in-place on the attn tensor in d_out).
// One block per (b,h,i) row of 1024 scores; re-zero masked entries afterward
// to match reference's where(mask==0, 0, attn).
// ---------------------------------------------------------------------------
__global__ __launch_bounds__(256) void softmax_mask_kernel(
    float* __restrict__ attn, const int* __restrict__ mask) {
  const int row = blockIdx.x;          // (b*NHEAD + h)*SEQ + i
  const int i   = row & (SEQ - 1);
  const int b   = row >> 14;           // row / (NHEAD*SEQ)
  float* rp = attn + (size_t)row * SEQ;
  const int* mp = mask + ((size_t)b * SEQ + i) * SEQ;

  __shared__ float red[256];
  const int tid = threadIdx.x;

  float v[4];
  float lmax = -3.0e38f;
#pragma unroll
  for (int t = 0; t < 4; ++t) {
    v[t] = rp[tid + t * 256];
    lmax = fmaxf(lmax, v[t]);
  }
  red[tid] = lmax;
  __syncthreads();
  for (int s = 128; s > 0; s >>= 1) {
    if (tid < s) red[tid] = fmaxf(red[tid], red[tid + s]);
    __syncthreads();
  }
  const float rmax = red[0];
  __syncthreads();

  float e[4];
  float lsum = 0.f;
#pragma unroll
  for (int t = 0; t < 4; ++t) {
    e[t] = __expf(v[t] - rmax);
    lsum += e[t];
  }
  red[tid] = lsum;
  __syncthreads();
  for (int s = 128; s > 0; s >>= 1) {
    if (tid < s) red[tid] += red[tid + s];
    __syncthreads();
  }
  const float inv = 1.0f / red[0];
#pragma unroll
  for (int t = 0; t < 4; ++t) {
    const int idx = tid + t * 256;
    rp[idx] = (mp[idx] == 0) ? 0.0f : e[t] * inv;
  }
}

// ---------------------------------------------------------------------------
// Batched 32x32 tiled transpose (+ convert) to bf16:  in (R x C) -> out (C x R)
// All dims here are multiples of 32, so no bounds checks.
// ---------------------------------------------------------------------------
template <typename T>
__global__ __launch_bounds__(256) void transpose_bf16_kernel(
    const T* __restrict__ in, __bf16* __restrict__ out,
    int R, int C, long inStride, long outStride) {
  __shared__ __bf16 tile[32][33];
  const T*     ib = in  + (size_t)blockIdx.z * inStride;
  __bf16*      ob = out + (size_t)blockIdx.z * outStride;
  const int c = blockIdx.x * 32 + threadIdx.x;
#pragma unroll
  for (int j = 0; j < 32; j += 8) {
    const int rr = blockIdx.y * 32 + threadIdx.y + j;
    tile[threadIdx.y + j][threadIdx.x] = (__bf16)ib[(size_t)rr * C + c];
  }
  __syncthreads();
  const int oc = blockIdx.y * 32 + threadIdx.x;  // out col = original row
#pragma unroll
  for (int j = 0; j < 32; j += 8) {
    const int orow = blockIdx.x * 32 + threadIdx.y + j;  // out row = original col
    ob[(size_t)orow * R + oc] = tile[threadIdx.x][threadIdx.y + j];
  }
}

// f32 -> bf16 elementwise (activations)
__global__ __launch_bounds__(256) void cvt_f32_bf16_kernel(
    const float* __restrict__ in, __bf16* __restrict__ out, size_t n) {
  size_t i = (size_t)blockIdx.x * blockDim.x + threadIdx.x;
  const size_t stride = (size_t)gridDim.x * blockDim.x;
  for (; i < n; i += stride) out[i] = (__bf16)in[i];
}

// ---------------------------------------------------------------------------
// Host driver
// ---------------------------------------------------------------------------
extern "C" void kernel_launch(void* const* d_in, const int* in_sizes, int n_in,
                              void* d_out, int out_size, void* d_ws, size_t ws_size,
                              hipStream_t stream) {
  (void)in_sizes; (void)n_in; (void)out_size; (void)ws_size;

  const float* query = (const float*)d_in[0];
  const float* key   = (const float*)d_in[1];
  const float* value = (const float*)d_in[2];
  const int*   mask  = (const int*)d_in[3];
  const float* Wq = (const float*)d_in[4];
  const float* bq = (const float*)d_in[5];
  const float* Wk = (const float*)d_in[6];
  const float* bk = (const float*)d_in[7];
  const float* Wv = (const float*)d_in[8];
  const float* bv = (const float*)d_in[9];
  const float* Wo = (const float*)d_in[10];
  const float* bo = (const float*)d_in[11];

  const int M  = BATCH * SEQ;          // 8192 rows for the big GEMMs
  const int BH = BATCH * NHEAD;        // 128 attention batches

  // ---- workspace carve (bf16 staging), ~136 MB total -----------------------
  char*  wsb = (char*)d_ws;
  size_t off = 0;
  auto carve = [&](size_t bytes) -> void* {
    void* p = wsb + off;
    off += (bytes + 255) & ~(size_t)255;
    return p;
  };
  __bf16* xq  = (__bf16*)carve((size_t)M * DMODEL * 2);
  __bf16* xk  = (__bf16*)carve((size_t)M * DMODEL * 2);
  __bf16* xv  = (__bf16*)carve((size_t)M * DMODEL * 2);
  __bf16* WqT = (__bf16*)carve((size_t)DMODEL * DMODEL * 2);
  __bf16* WkT = (__bf16*)carve((size_t)DMODEL * DMODEL * 2);
  __bf16* WvT = (__bf16*)carve((size_t)DMODEL * DMODEL * 2);
  __bf16* WoT = (__bf16*)carve((size_t)DMODEL * DMODEL * 2);
  __bf16* Qb  = (__bf16*)carve((size_t)M * DMODEL * 2);  // flat == (B,H,S,DK) view
  __bf16* Kb  = (__bf16*)carve((size_t)M * DMODEL * 2);
  __bf16* Vb  = (__bf16*)carve((size_t)M * DMODEL * 2);
  __bf16* Vt  = (__bf16*)carve((size_t)BH * HDIM * SEQ * 2);  // per (b,h): (DV x S)
  __bf16* ctx = (__bf16*)carve((size_t)M * DMODEL * 2);  // flat == (B,S,H*DV) view

  float* out_p  = (float*)d_out;
  float* attn_p = (float*)d_out + (size_t)BATCH * SEQ * DMODEL;

  // ---- 1) convert activations, transpose+convert weights -------------------
  const size_t nElem = (size_t)M * DMODEL;
  cvt_f32_bf16_kernel<<<2048, 256, 0, stream>>>(query, xq, nElem);
  cvt_f32_bf16_kernel<<<2048, 256, 0, stream>>>(key,   xk, nElem);
  cvt_f32_bf16_kernel<<<2048, 256, 0, stream>>>(value, xv, nElem);

  dim3 tb(32, 8);
  transpose_bf16_kernel<float><<<dim3(32, 32, 1), tb, 0, stream>>>(Wq, WqT, DMODEL, DMODEL, 0, 0);
  transpose_bf16_kernel<float><<<dim3(32, 32, 1), tb, 0, stream>>>(Wk, WkT, DMODEL, DMODEL, 0, 0);
  transpose_bf16_kernel<float><<<dim3(32, 32, 1), tb, 0, stream>>>(Wv, WvT, DMODEL, DMODEL, 0, 0);
  transpose_bf16_kernel<float><<<dim3(32, 32, 1), tb, 0, stream>>>(Wo, WoT, DMODEL, DMODEL, 0, 0);

  // ---- 2) QKV projections: (8192x1024) x (1024x1024) + bias -> bf16 --------
  dim3 gProj(DMODEL / 64, M / 64, 1);
  gemm_wmma_kernel<__bf16, 1><<<gProj, 128, 0, stream>>>(
      xq, WqT, Qb, bq, nullptr, DMODEL, DMODEL, DMODEL, DMODEL, 0, 0, 0, 1.f);
  gemm_wmma_kernel<__bf16, 1><<<gProj, 128, 0, stream>>>(
      xk, WkT, Kb, bk, nullptr, DMODEL, DMODEL, DMODEL, DMODEL, 0, 0, 0, 1.f);
  gemm_wmma_kernel<__bf16, 1><<<gProj, 128, 0, stream>>>(
      xv, WvT, Vb, bv, nullptr, DMODEL, DMODEL, DMODEL, DMODEL, 0, 0, 0, 1.f);

  // ---- 3) V^T per head: (S x DV) -> (DV x S), 128 batches ------------------
  transpose_bf16_kernel<__bf16><<<dim3(HDIM / 32, SEQ / 32, BH), tb, 0, stream>>>(
      Vb, Vt, SEQ, HDIM, (long)SEQ * HDIM, (long)HDIM * SEQ);

  // ---- 4) scores = (Q/DK) K^T, masked -> attn region of d_out --------------
  dim3 gScore(SEQ / 64, SEQ / 64, BH);
  gemm_wmma_kernel<__bf16, 2><<<gScore, 128, 0, stream>>>(
      Qb, Kb, attn_p, nullptr, mask, HDIM, HDIM, HDIM, SEQ,
      (long)SEQ * HDIM, (long)SEQ * HDIM, (long)SEQ * SEQ, 1.0f / HDIM);

  // ---- 5) softmax + mask-zero, in place on attn ----------------------------
  softmax_mask_kernel<<<BH * SEQ, 256, 0, stream>>>(attn_p, mask);

  // ---- 6) context = attn (f32) x V  -> bf16 flat (B,H,S,DV) ----------------
  dim3 gCtx(HDIM / 64, SEQ / 64, BH);
  gemm_wmma_kernel<float, 1><<<gCtx, 128, 0, stream>>>(
      attn_p, Vt, ctx, nullptr, nullptr, SEQ, SEQ, SEQ, HDIM,
      (long)SEQ * SEQ, (long)HDIM * SEQ, (long)SEQ * HDIM, 1.f);

  // ---- 7) out = ctx x Wo + bo -> f32 d_out ---------------------------------
  gemm_wmma_kernel<__bf16, 0><<<gProj, 128, 0, stream>>>(
      ctx, WoT, out_p, bo, nullptr, DMODEL, DMODEL, DMODEL, DMODEL, 0, 0, 0, 1.f);
}
